// GENnet_6717328851287
// MI455X (gfx1250) — matrix operand order repaired and checked by reading, not compiled, not run on hardware
//
#include <hip/hip_runtime.h>

#define NN 50000
#define EE 800000
#define GG 128
#define DD 128
#define MLPH 256
#define LAYERS 4

typedef __attribute__((ext_vector_type(16))) __bf16 v16bf;
typedef __attribute__((ext_vector_type(8)))  float  v8f;
typedef __attribute__((ext_vector_type(4)))  unsigned int v4u;
typedef __attribute__((ext_vector_type(8)))  int v8i;
typedef __attribute__((ext_vector_type(4)))  int v4i;

union FragB { v16bf v; uint4 q[2]; };

#if defined(__has_builtin)
#if __has_builtin(__builtin_amdgcn_tensor_load_to_lds) && __has_builtin(__builtin_amdgcn_s_wait_tensorcnt)
#define HAVE_TDM 1
#endif
#endif

#if defined(HAVE_TDM)
#if __has_include(<hip/amd_detail/amd_gfx1250_TDM.h>)
// amdgpu-toolchain (clang-23): 6-arg form
#define TDM_LOAD(g0, g1, g2, g3) \
  __builtin_amdgcn_tensor_load_to_lds((g0), (g1), (g2), (g3), (v8i){0,0,0,0,0,0,0,0}, 0)
#else
// ROCm 7.2 (clang-22): 5-arg form
#define TDM_LOAD(g0, g1, g2, g3) \
  __builtin_amdgcn_tensor_load_to_lds((g0), (g1), (g2), (g3), 0)
#endif
#endif

__device__ __forceinline__ unsigned short f2bf(float f) {
  unsigned u = __float_as_uint(f);
  unsigned r = u + 0x7fffu + ((u >> 16) & 1u);   // round-to-nearest-even
  return (unsigned short)(r >> 16);
}
__device__ __forceinline__ float bf2f(unsigned short s) {
  return __uint_as_float(((unsigned)s) << 16);
}
// order-preserving float->uint so unsigned atomicMax == float max
__device__ __forceinline__ unsigned ordf(float f) {
  unsigned u = __float_as_uint(f);
  return (u & 0x80000000u) ? ~u : (u | 0x80000000u);
}
__device__ __forceinline__ float unordf(unsigned o) {
  return (o & 0x80000000u) ? __uint_as_float(o & 0x7fffffffu) : __uint_as_float(~o);
}

// ---------------- utility kernels ----------------
__global__ void fill_zero_kernel(unsigned* p, size_t n) {
  size_t i = (size_t)blockIdx.x * blockDim.x + threadIdx.x;
  if (i < n) p[i] = 0u;
}

// fp32 [M,K0] -> bf16 [M,Kp] (zero pad K0..Kp)
__global__ void cvt_a_kernel(const float* __restrict__ A, unsigned short* __restrict__ O,
                             int M, int K0, int Kp) {
  size_t i = (size_t)blockIdx.x * blockDim.x + threadIdx.x;
  size_t total = (size_t)M * Kp;
  if (i >= total) return;
  int k = (int)(i % Kp);
  size_t r = i / Kp;
  float v = (k < K0) ? A[r * K0 + k] : 0.f;
  O[i] = f2bf(v);
}

// fp32 W[K0,Nc] -> bf16 Wt[Nc,Kp] (transposed, zero pad rows K0..Kp)
__global__ void cvt_w_kernel(const float* __restrict__ W, unsigned short* __restrict__ Wt,
                             int K0, int Kp, int Nc) {
  size_t i = (size_t)blockIdx.x * blockDim.x + threadIdx.x;
  size_t total = (size_t)Nc * Kp;
  if (i >= total) return;
  int k = (int)(i % Kp);
  size_t n = i / Kp;
  float v = (k < K0) ? W[(size_t)k * Nc + n] : 0.f;
  Wt[i] = f2bf(v);
}

// ---------------- WMMA GEMM: C[M,N] = A[M,K](bf16) * Wt[N,K](bf16)^T + bias ----------------
// block = 256 threads (8 waves); each wave computes a 16x64 tile (4 accumulators).
// Weight panel Wt[0:N, kc:kc+Kc] is staged into a 64KB LDS buffer via the Tensor Data
// Mover (tensor_load_to_lds + s_wait_tensorcnt), then fed to WMMA via ds loads.
// Requires: N % 64 == 0, K % 32 == 0, K % Kc == 0 with Kc = min(K, 32768/N).
__global__ __launch_bounds__(256)
void wmma_gemm_kernel(const unsigned short* __restrict__ A,
                      const unsigned short* __restrict__ Wt,
                      const float* __restrict__ bias,
                      float* __restrict__ Cf,
                      unsigned short* __restrict__ Cb,
                      int M, int N, int K, int relu) {
  __shared__ unsigned short WLDS[32768];            // 64 KB weight panel

  const int wave = threadIdx.x >> 5;
  const int lane = threadIdx.x & 31;
  const int wpn  = N >> 6;                  // waves per N (N/64)
  const int nw   = wave % wpn;
  const int rw   = wave / wpn;
  const int rt   = 8 / wpn;                 // row tiles per block
  const int rowTile = blockIdx.x * rt + rw;
  const bool active = (rowTile * 16) < M;   // wave-uniform
  const int row0 = rowTile * 16;
  const int col0 = nw * 64;

  const int r    = lane & 15;
  const int hk   = (lane >> 4) << 3;        // 0 or 8 (A layout)
  const int bkof = (lane >> 4) << 4;        // 0 or 16 (unused; B uses hk too)
  const int bc   = col0 + (lane & 15);      // B column for this lane

  const int kcMax = 32768 / N;
  const int Kc = (K < kcMax) ? K : kcMax;   // LDS panel depth (multiple of 32)

  v8f acc[4];
  acc[0] = (v8f){}; acc[1] = (v8f){}; acc[2] = (v8f){}; acc[3] = (v8f){};

  const unsigned short* arow = A + (size_t)(row0 + r) * K;

  for (int kc = 0; kc < K; kc += Kc) {
#if defined(HAVE_TDM)
    if (threadIdx.x < 32) {                 // one wave drives the TDM
      unsigned long long ga = (unsigned long long)(size_t)Wt + (unsigned long long)kc * 2ull;
      unsigned lva = (unsigned)(size_t)&WLDS[0];          // LDS byte offset
      v4u g0;
      g0.x = 1u;                                          // count=1, load descriptor
      g0.y = lva;                                         // lds_addr
      g0.z = (unsigned)ga;                                // global_addr[31:0]
      g0.w = ((unsigned)(ga >> 32) & 0x01ffffffu) | (2u << 30);  // addr[56:32] | type=2
      v8i g1;
      g1[0] = (int)(1u << 16);                            // wg_mask=0, data_size=1 (2B)
      g1[1] = (int)(((unsigned)K & 0xffffu) << 16);       // tensor_dim0 lo16 (bits 63:48)
      g1[2] = (int)((((unsigned)K >> 16) & 0xffffu) |     // tensor_dim0 hi16
                    (((unsigned)N & 0xffffu) << 16));     // tensor_dim1 lo16
      g1[3] = (int)((((unsigned)N >> 16) & 0xffffu) |     // tensor_dim1 hi16
                    (((unsigned)Kc & 0xffffu) << 16));    // tile_dim0 = Kc
      g1[4] = (int)((unsigned)N & 0xffffu);               // tile_dim1 = N, tile_dim2 = 0
      g1[5] = (int)(unsigned)K;                           // tensor_dim0_stride lo32
      g1[6] = 0;                                          // stride0 hi16, stride1 lo16
      g1[7] = 0;
      v4i gz = {0, 0, 0, 0};
      TDM_LOAD(g0, g1, gz, gz);
      __builtin_amdgcn_s_wait_tensorcnt(0);
    }
    __syncthreads();
#else
    // fallback: cooperative global->LDS copy (16B per thread-iteration)
    for (int t = threadIdx.x; t < (N * Kc) / 8; t += 256) {
      int row  = (t * 8) / Kc;
      int colk = (t * 8) % Kc;
      *(uint4*)&WLDS[t * 8] = *(const uint4*)(Wt + (size_t)row * K + kc + colk);
    }
    __syncthreads();
#endif

    if (active) {
      __builtin_prefetch(arow + kc + Kc, 0, 0);   // speculative prefetch of next A chunk
      for (int kt = 0; kt < Kc; kt += 32) {
        FragB af;
        af.q[0] = *(const uint4*)(arow + kc + kt + hk);        // K = +hk .. +7
        af.q[1] = *(const uint4*)(arow + kc + kt + hk + 16);   // K = +hk+16 .. +23
#pragma unroll
        for (int j = 0; j < 4; ++j) {
          const unsigned short* wr = &WLDS[(size_t)(bc + j * 16) * Kc + kt + hk];
          FragB bfr;
          bfr.q[0] = *(const uint4*)(wr);
          bfr.q[1] = *(const uint4*)(wr + 16);
          acc[j] = __builtin_amdgcn_wmma_f32_16x16x32_bf16(
              false, af.v, false, bfr.v, (short)0, acc[j], false, false);
        }
      }
    }
    __syncthreads();   // panel consumed before next chunk overwrites LDS
  }

  if (active) {
    const int rbase = row0 + ((lane >> 4) << 3);
#pragma unroll
    for (int j = 0; j < 4; ++j) {
      int c = col0 + j * 16 + (lane & 15);
      float bv = bias ? bias[c] : 0.f;
#pragma unroll
      for (int i = 0; i < 8; ++i) {
        float v = acc[j][i] + bv;
        if (relu) v = fmaxf(v, 0.f);
        size_t off = (size_t)(rbase + i) * N + c;
        if (Cf) Cf[off] = v;
        if (Cb) Cb[off] = f2bf(v);
      }
    }
  }
}

// ---------------- edge phase ----------------
// pass 1: per (edge, 4 channels): msg = relu(h[src]+ea)+eps; atomicMax of msg*t into MB[dst]
__global__ void edge_max_kernel(const int* __restrict__ src, const int* __restrict__ dst,
                                const float* __restrict__ HF,
                                const unsigned short* __restrict__ EABF,
                                const float* __restrict__ T, int l,
                                unsigned* __restrict__ MB) {
  size_t i = (size_t)blockIdx.x * blockDim.x + threadIdx.x;
  if (i >= (size_t)EE * 32) return;
  int e  = (int)(i >> 5);
  int cg = ((int)i & 31) << 2;
  int s = src[e], d = dst[e];
  float4  h  = *(const float4*)(HF + (size_t)s * DD + cg);
  ushort4 ea = *(const ushort4*)(EABF + (size_t)e * DD + cg);
  float t = T[l];
  float m0 = fmaxf(h.x + bf2f(ea.x), 0.f) + 1e-7f;
  float m1 = fmaxf(h.y + bf2f(ea.y), 0.f) + 1e-7f;
  float m2 = fmaxf(h.z + bf2f(ea.z), 0.f) + 1e-7f;
  float m3 = fmaxf(h.w + bf2f(ea.w), 0.f) + 1e-7f;
  unsigned* mb = MB + (size_t)d * DD + cg;
  atomicMax(mb + 0, ordf(m0 * t));
  atomicMax(mb + 1, ordf(m1 * t));
  atomicMax(mb + 2, ordf(m2 * t));
  atomicMax(mb + 3, ordf(m3 * t));
}

// pass 2: e = exp(msg*t - max); SB += e; WB += e*msg
__global__ void edge_sum_kernel(const int* __restrict__ src, const int* __restrict__ dst,
                                const float* __restrict__ HF,
                                const unsigned short* __restrict__ EABF,
                                const float* __restrict__ T, int l,
                                const unsigned* __restrict__ MB,
                                float* __restrict__ SB, float* __restrict__ WB) {
  size_t i = (size_t)blockIdx.x * blockDim.x + threadIdx.x;
  if (i >= (size_t)EE * 32) return;
  int e  = (int)(i >> 5);
  int cg = ((int)i & 31) << 2;
  int s = src[e], d = dst[e];
  float4  h  = *(const float4*)(HF + (size_t)s * DD + cg);
  ushort4 ea = *(const ushort4*)(EABF + (size_t)e * DD + cg);
  float t = T[l];
  float m0 = fmaxf(h.x + bf2f(ea.x), 0.f) + 1e-7f;
  float m1 = fmaxf(h.y + bf2f(ea.y), 0.f) + 1e-7f;
  float m2 = fmaxf(h.z + bf2f(ea.z), 0.f) + 1e-7f;
  float m3 = fmaxf(h.w + bf2f(ea.w), 0.f) + 1e-7f;
  size_t off = (size_t)d * DD + cg;
  float e0 = __expf(m0 * t - unordf(MB[off + 0]));
  float e1 = __expf(m1 * t - unordf(MB[off + 1]));
  float e2 = __expf(m2 * t - unordf(MB[off + 2]));
  float e3 = __expf(m3 * t - unordf(MB[off + 3]));
  atomicAdd(SB + off + 0, e0);  atomicAdd(WB + off + 0, e0 * m0);
  atomicAdd(SB + off + 1, e1);  atomicAdd(WB + off + 1, e1 * m1);
  atomicAdd(SB + off + 2, e2);  atomicAdd(WB + off + 2, e2 * m2);
  atomicAdd(SB + off + 3, e3);  atomicAdd(WB + off + 3, e3 * m3);
}

// agg = WB/(SB+1e-16); hres = agg + h ; store bf16 for GEMM
__global__ void node_update_kernel(const float* __restrict__ HF,
                                   const float* __restrict__ SB,
                                   const float* __restrict__ WB,
                                   unsigned short* __restrict__ HRBF) {
  size_t i = (size_t)blockIdx.x * blockDim.x + threadIdx.x;
  if (i >= (size_t)NN * DD) return;
  float agg = WB[i] / (SB[i] + 1e-16f);
  HRBF[i] = f2bf(agg + HF[i]);
}

// ---------------- BatchNorm ----------------
__global__ void colstats_kernel(const float* __restrict__ Y, float* __restrict__ stats,
                                int M, int Nc) {
  int c = threadIdx.x;          // blockDim.x == Nc
  float s = 0.f, q = 0.f;
  for (int r = blockIdx.x; r < M; r += gridDim.x) {
    float v = Y[(size_t)r * Nc + c];
    s += v; q += v * v;
  }
  atomicAdd(stats + c, s);
  atomicAdd(stats + Nc + c, q);
}

__global__ void bn_relu_kernel(const float* __restrict__ Y, const float* __restrict__ stats,
                               const float* __restrict__ g, const float* __restrict__ b,
                               int M, int Nc, float* __restrict__ outF,
                               unsigned short* __restrict__ outB) {
  size_t i = (size_t)blockIdx.x * blockDim.x + threadIdx.x;
  if (i >= (size_t)M * Nc) return;
  int c = (int)(i % Nc);
  float invM = 1.f / (float)M;
  float mu  = stats[c] * invM;
  float var = stats[Nc + c] * invM - mu * mu;
  float v = (Y[i] - mu) * rsqrtf(var + 1e-5f) * g[c] + b[c];
  v = fmaxf(v, 0.f);
  if (outF) outF[i] = v;
  if (outB) outB[i] = f2bf(v);
}

// ---------------- pooling + head ----------------
__global__ void pool_sum_kernel(const float* __restrict__ HF, const int* __restrict__ batch,
                                float* __restrict__ POOL, float* __restrict__ CNT) {
  size_t i = (size_t)blockIdx.x * blockDim.x + threadIdx.x;
  if (i >= (size_t)NN * DD) return;
  int n = (int)(i >> 7);
  int c = (int)(i & 127);
  int g = batch[n];
  atomicAdd(POOL + (size_t)g * DD + c, HF[i]);
  if (c == 0) atomicAdd(CNT + g, 1.f);
}

__global__ void pool_div_kernel(const float* __restrict__ POOL, const float* __restrict__ CNT,
                                unsigned short* __restrict__ POOLBF) {
  int i = blockIdx.x * blockDim.x + threadIdx.x;
  if (i >= GG * DD) return;
  int g = i >> 7;
  POOLBF[i] = f2bf(POOL[i] / fmaxf(CNT[g], 1.f));
}

__global__ void head_final_kernel(const unsigned short* __restrict__ Z2BF,
                                  const float* __restrict__ fW, const float* __restrict__ fb,
                                  float* __restrict__ out) {
  int i = blockIdx.x * blockDim.x + threadIdx.x;
  if (i >= GG * 10) return;
  int g = i / 10, o = i % 10;
  float acc = fb[o];
  const unsigned short* z = Z2BF + (size_t)g * 256;
  for (int k = 0; k < 256; ++k) acc += bf2f(z[k]) * fW[k * 10 + o];
  out[i] = acc;
}

// ---------------- host ----------------
extern "C" void kernel_launch(void* const* d_in, const int* in_sizes, int n_in,
                              void* d_out, int out_size, void* d_ws, size_t ws_size,
                              hipStream_t stream) {
  const float* x         = (const float*)d_in[0];
  const int*   eidx      = (const int*)  d_in[1];
  const float* edge_attr = (const float*)d_in[2];
  const int*   batch     = (const int*)  d_in[3];
  const float* node_W    = (const float*)d_in[4];
  const float* node_b    = (const float*)d_in[5];
  const float* edge_W    = (const float*)d_in[6];
  const float* edge_b    = (const float*)d_in[7];
  const float* conv_t    = (const float*)d_in[8];
  const float* conv_w1   = (const float*)d_in[9];
  const float* conv_b1   = (const float*)d_in[10];
  const float* conv_bng  = (const float*)d_in[11];
  const float* conv_bnb  = (const float*)d_in[12];
  const float* conv_w2   = (const float*)d_in[13];
  const float* conv_b2   = (const float*)d_in[14];
  const float* bn_g      = (const float*)d_in[15];
  const float* bn_b      = (const float*)d_in[16];
  const float* d1_W      = (const float*)d_in[17];
  const float* d1_b      = (const float*)d_in[18];
  const float* d2_W      = (const float*)d_in[19];
  const float* d2_b      = (const float*)d_in[20];
  const float* f_W       = (const float*)d_in[21];
  const float* f_b       = (const float*)d_in[22];
  (void)in_sizes; (void)n_in; (void)out_size; (void)ws_size;

  const int* src = eidx;
  const int* dst = eidx + EE;

  char* base = (char*)d_ws;
  size_t off = 0;
  auto carve = [&](size_t bytes) -> char* {
    off = (off + 255) & ~(size_t)255;
    char* p = base + off;
    off += bytes;
    return p;
  };
  unsigned short* XBF    = (unsigned short*)carve((size_t)NN * 64 * 2);
  unsigned short* EAIN   = (unsigned short*)carve((size_t)EE * 32 * 2);
  unsigned short* EABF   = (unsigned short*)carve((size_t)EE * DD * 2);
  float*          HF     = (float*)carve((size_t)NN * DD * 4);
  unsigned*       MB     = (unsigned*)carve((size_t)NN * DD * 4);  // MB,SB,WB contiguous
  float*          SB     = (float*)carve((size_t)NN * DD * 4);
  float*          WB     = (float*)carve((size_t)NN * DD * 4);
  unsigned short* HRBF   = (unsigned short*)carve((size_t)NN * DD * 2);
  float*          Y      = (float*)carve((size_t)NN * MLPH * 4);
  unsigned short* YBF    = (unsigned short*)carve((size_t)NN * MLPH * 2);
  float*          Z      = (float*)carve((size_t)NN * DD * 4);
  float*          STATS  = (float*)carve((size_t)2 * 512 * 4);
  float*          POOL   = (float*)carve((size_t)GG * DD * 4);     // POOL,CNT contiguous
  float*          CNT    = (float*)carve((size_t)GG * 4);
  unsigned short* POOLBF = (unsigned short*)carve((size_t)GG * DD * 2);
  unsigned short* Z1BF   = (unsigned short*)carve((size_t)GG * 512 * 2);
  unsigned short* Z2BF   = (unsigned short*)carve((size_t)GG * 256 * 2);
  unsigned short* NODEWT = (unsigned short*)carve((size_t)DD * 64 * 2);
  unsigned short* EDGEWT = (unsigned short*)carve((size_t)DD * 32 * 2);
  unsigned short* W1T    = (unsigned short*)carve((size_t)LAYERS * MLPH * DD * 2);
  unsigned short* W2T    = (unsigned short*)carve((size_t)LAYERS * DD * MLPH * 2);
  unsigned short* D1T    = (unsigned short*)carve((size_t)512 * DD * 2);
  unsigned short* D2T    = (unsigned short*)carve((size_t)256 * 512 * 2);

  auto nblk = [](size_t n, int b) { return (unsigned)((n + (size_t)b - 1) / (size_t)b); };

  // input/weight conversions to bf16 (weights transposed to [N,K])
  cvt_a_kernel<<<nblk((size_t)NN * 64, 256), 256, 0, stream>>>(x, XBF, NN, 64, 64);
  cvt_a_kernel<<<nblk((size_t)EE * 32, 256), 256, 0, stream>>>(edge_attr, EAIN, EE, 16, 32);
  cvt_w_kernel<<<nblk((size_t)128 * 64, 256), 256, 0, stream>>>(node_W, NODEWT, 64, 64, 128);
  cvt_w_kernel<<<nblk((size_t)128 * 32, 256), 256, 0, stream>>>(edge_W, EDGEWT, 16, 32, 128);
  for (int l = 0; l < LAYERS; ++l) {
    cvt_w_kernel<<<nblk((size_t)MLPH * DD, 256), 256, 0, stream>>>(
        conv_w1 + (size_t)l * DD * MLPH, W1T + (size_t)l * MLPH * DD, DD, DD, MLPH);
    cvt_w_kernel<<<nblk((size_t)DD * MLPH, 256), 256, 0, stream>>>(
        conv_w2 + (size_t)l * MLPH * DD, W2T + (size_t)l * DD * MLPH, MLPH, MLPH, DD);
  }
  cvt_w_kernel<<<nblk((size_t)512 * DD, 256), 256, 0, stream>>>(d1_W, D1T, 128, 128, 512);
  cvt_w_kernel<<<nblk((size_t)256 * 512, 256), 256, 0, stream>>>(d2_W, D2T, 512, 512, 256);

  auto gemm = [&](const unsigned short* A, const unsigned short* Wt, const float* bias,
                  float* Cf, unsigned short* Cb, int M, int N, int K, int relu) {
    int wpn = N / 64;
    int rt  = 8 / wpn;
    int rowTiles = (M + 15) / 16;
    int grid = (rowTiles + rt - 1) / rt;
    wmma_gemm_kernel<<<grid, 256, 0, stream>>>(A, Wt, bias, Cf, Cb, M, N, K, relu);
  };

  // h0 = x @ node_W + node_b (fp32); ea = pad(edge_attr) @ edge_W + edge_b (bf16)
  gemm(XBF, NODEWT, node_b, HF, nullptr, NN, 128, 64, 0);
  gemm(EAIN, EDGEWT, edge_b, nullptr, EABF, EE, 128, 32, 0);

  for (int l = 0; l < LAYERS; ++l) {
    // zero MB/SB/WB (contiguous: sizes are 256B multiples)
    fill_zero_kernel<<<nblk((size_t)3 * NN * DD, 256), 256, 0, stream>>>(MB, (size_t)3 * NN * DD);
    edge_max_kernel<<<nblk((size_t)EE * 32, 256), 256, 0, stream>>>(src, dst, HF, EABF, conv_t, l, MB);
    edge_sum_kernel<<<nblk((size_t)EE * 32, 256), 256, 0, stream>>>(src, dst, HF, EABF, conv_t, l, MB, SB, WB);
    node_update_kernel<<<nblk((size_t)NN * DD, 256), 256, 0, stream>>>(HF, SB, WB, HRBF);

    gemm(HRBF, W1T + (size_t)l * MLPH * DD, conv_b1 + (size_t)l * MLPH, Y, nullptr, NN, MLPH, DD, 0);
    fill_zero_kernel<<<nblk((size_t)2 * MLPH, 256), 256, 0, stream>>>((unsigned*)STATS, (size_t)2 * MLPH);
    colstats_kernel<<<512, MLPH, 0, stream>>>(Y, STATS, NN, MLPH);
    bn_relu_kernel<<<nblk((size_t)NN * MLPH, 256), 256, 0, stream>>>(
        Y, STATS, conv_bng + (size_t)l * MLPH, conv_bnb + (size_t)l * MLPH, NN, MLPH, nullptr, YBF);

    gemm(YBF, W2T + (size_t)l * DD * MLPH, conv_b2 + (size_t)l * DD, Z, nullptr, NN, DD, MLPH, 0);
    fill_zero_kernel<<<nblk((size_t)2 * DD, 256), 256, 0, stream>>>((unsigned*)STATS, (size_t)2 * DD);
    colstats_kernel<<<512, DD, 0, stream>>>(Z, STATS, NN, DD);
    bn_relu_kernel<<<nblk((size_t)NN * DD, 256), 256, 0, stream>>>(
        Z, STATS, bn_g + (size_t)l * DD, bn_b + (size_t)l * DD, NN, DD, HF, nullptr);
  }

  // global mean pool
  fill_zero_kernel<<<nblk((size_t)GG * DD + GG, 256), 256, 0, stream>>>((unsigned*)POOL, (size_t)GG * DD + GG);
  pool_sum_kernel<<<nblk((size_t)NN * DD, 256), 256, 0, stream>>>(HF, batch, POOL, CNT);
  pool_div_kernel<<<nblk((size_t)GG * DD, 256), 256, 0, stream>>>(POOL, CNT, POOLBF);

  // head: relu(pooled@d1+b) -> relu(@d2+b) -> @f+b
  gemm(POOLBF, D1T, d1_b, nullptr, Z1BF, GG, 512, 128, 1);
  gemm(Z1BF, D2T, d2_b, nullptr, Z2BF, GG, 256, 512, 1);
  head_final_kernel<<<nblk((size_t)GG * 10, 256), 256, 0, stream>>>(Z2BF, f_W, f_b, (float*)d_out);
}
